// MiniBatchSOM_60275571032373
// MI455X (gfx1250) — matrix operand order repaired and checked by reading, not compiled, not run on hardware
//
#include <hip/hip_runtime.h>
#include <hip/hip_bf16.h>

// ---------------------------------------------------------------------------
// MiniBatch SOM on MI455X (gfx1250, wave32, WMMA)
//   B=16384 samples, DIM=256, map 64x64 (K=4096)
// ---------------------------------------------------------------------------

typedef __bf16  v16bf __attribute__((ext_vector_type(16)));
typedef __bf16  v8bf  __attribute__((ext_vector_type(8)));
typedef float   v8f   __attribute__((ext_vector_type(8)));

#define SOM_B     16384
#define SOM_D     256
#define SOM_K     4096     // 64*64 map units
#define SOM_GW    64       // grid width (N)

__device__ __forceinline__ v8f wmma_bf16(v16bf a, v16bf b, v8f c) {
    // D = A(16x32 bf16) * B(32x16 bf16) + C(16x16 f32)
    return __builtin_amdgcn_wmma_f32_16x16x32_bf16(
        false, a, false, b, (short)0, c, false, false);
}

// truncated hi/lo split: v == (float)hi + (float)lo to ~16-17 mantissa bits
__device__ __forceinline__ void split_bf16(float v, __bf16* hi, __bf16* lo) {
    unsigned u = __float_as_uint(v) & 0xffff0000u;
    float fh = __uint_as_float(u);     // exactly representable in bf16
    *hi = (__bf16)fh;
    *lo = (__bf16)(v - fh);
}

// ---------------------------------------------------------------------------
// Kernel 1: split x into bf16 hi/lo (row-major [B][D])
// ---------------------------------------------------------------------------
__global__ void __launch_bounds__(256) som_prep_x(
    const float* __restrict__ x, __bf16* __restrict__ xh, __bf16* __restrict__ xl) {
    int g = blockIdx.x * 256 + threadIdx.x;
    __bf16 h, l;
    split_bf16(x[g], &h, &l);
    xh[g] = h;
    xl[g] = l;
}

// ---------------------------------------------------------------------------
// Kernel 2: split weights into TRANSPOSED bf16 hi/lo ([D][K], so the WMMA
// B-operand rows are contiguous) and compute ||w_j||^2
// ---------------------------------------------------------------------------
__global__ void __launch_bounds__(256) som_prep_w(
    const float* __restrict__ w, __bf16* __restrict__ whT, __bf16* __restrict__ wlT,
    float* __restrict__ wn) {
    __shared__ float red[256];
    const int j = blockIdx.x;       // map unit
    const int d = threadIdx.x;      // feature dim (== 256 threads)
    float v = w[(size_t)j * SOM_D + d];
    __bf16 h, l;
    split_bf16(v, &h, &l);
    whT[(size_t)d * SOM_K + j] = h;
    wlT[(size_t)d * SOM_K + j] = l;
    red[d] = v * v;
    __syncthreads();
    for (int s = 128; s > 0; s >>= 1) {
        if (d < s) red[d] += red[d + s];
        __syncthreads();
    }
    if (d == 0) wn[j] = red[0];
}

// ---------------------------------------------------------------------------
// Kernel 3: zero num/den partial-sum buffers (accum kernel uses atomics)
// ---------------------------------------------------------------------------
__global__ void __launch_bounds__(256) som_zero(
    float* __restrict__ num, float* __restrict__ den) {
    int g = blockIdx.x * 256 + threadIdx.x;
    if (g < SOM_K * SOM_D) num[g] = 0.0f;
    else                   den[g - SOM_K * SOM_D] = 0.0f;
}

// ---------------------------------------------------------------------------
// Kernel 4: BMU search. Wave owns 16 batch rows (A hi+lo cached in 128
// VGPRs); the block double-buffers each 16-KB w-tile (hi+lo) through LDS so
// the 8 waves share one L2 read instead of eight. 24 bf16 WMMAs per j-tile
// (hi*hi + hi*lo + lo*hi), fused argmin.
// score = ||w||^2 - 2 x.w  (||x||^2 constant under argmin)
// ---------------------------------------------------------------------------
__global__ void __launch_bounds__(256) som_bmu(
    const __bf16* __restrict__ xh, const __bf16* __restrict__ xl,
    const __bf16* __restrict__ whT, const __bf16* __restrict__ wlT,
    const float* __restrict__ wn,
    int* __restrict__ bmu_idx, float2* __restrict__ bmu_loc) {

    __shared__ __align__(32) __bf16 sWh[2][SOM_D * 16];   // [buf][d*16 + n]
    __shared__ __align__(32) __bf16 sWl[2][SOM_D * 16];

    const int tid  = threadIdx.x;
    const int lane = tid & 31;
    const int wave = tid >> 5;
    const int x0   = blockIdx.x * 128 + wave * 16;  // 16 rows per wave, 8 waves
    const int r    = lane & 15;
    const int hs   = lane >> 4;

    // ---- preload the wave's entire A (16 x 256, hi+lo) into registers ----
    // 16-bit A layout: lane<16 -> M=lane, K = c*32 + [0..7] and [16..23]
    //                  lane>=16 -> M=lane-16, K = c*32 + [8..15] and [24..31]
    v16bf Ah[8], Al[8];
    {
        const __bf16* xrh = xh + (size_t)(x0 + r) * SOM_D + 8 * hs;
        const __bf16* xrl = xl + (size_t)(x0 + r) * SOM_D + 8 * hs;
#pragma unroll
        for (int c = 0; c < 8; ++c) {
            v8bf s0 = *(const v8bf*)(xrh + c * 32);
            v8bf s1 = *(const v8bf*)(xrh + c * 32 + 16);
            Ah[c] = __builtin_shufflevector(s0, s1, 0,1,2,3,4,5,6,7,8,9,10,11,12,13,14,15);
            v8bf t0 = *(const v8bf*)(xrl + c * 32);
            v8bf t1 = *(const v8bf*)(xrl + c * 32 + 16);
            Al[c] = __builtin_shufflevector(t0, t1, 0,1,2,3,4,5,6,7,8,9,10,11,12,13,14,15);
        }
    }

    // ---- LDS staging helpers: thread t owns d-row t of the w-tile ----
    const __bf16* gWh = whT + (size_t)tid * SOM_K;
    const __bf16* gWl = wlT + (size_t)tid * SOM_K;
    uint4 rh0, rh1, rl0, rl1;
    auto stage_load = [&](int jt) {
        const uint4* ph = (const uint4*)(gWh + jt * 16);
        const uint4* pl = (const uint4*)(gWl + jt * 16);
        rh0 = ph[0]; rh1 = ph[1];
        rl0 = pl[0]; rl1 = pl[1];
    };
    auto stage_store = [&](int buf) {
        uint4* qh = (uint4*)&sWh[buf][tid * 16];
        uint4* ql = (uint4*)&sWl[buf][tid * 16];
        qh[0] = rh0; qh[1] = rh1;
        ql[0] = rl0; ql[1] = rl1;
    };

    float bestV[8];
    int   bestI[8];
#pragma unroll
    for (int q = 0; q < 8; ++q) { bestV[q] = 3.4e38f; bestI[q] = 0; }

    stage_load(0);
    stage_store(0);

    for (int jt = 0; jt < SOM_K / 16; ++jt) {
        const int buf = jt & 1;
        __syncthreads();                      // tile `buf` visible to all waves
        if (jt + 1 < SOM_K / 16) {
            stage_load(jt + 1);               // global loads hide under WMMAs
            if (jt + 2 < SOM_K / 16) {
                __builtin_prefetch(gWh + (jt + 2) * 16, 0, 3);
                __builtin_prefetch(gWl + (jt + 2) * 16, 0, 3);
            }
        }

        const int j0 = jt * 16;
        v8f C = {0.f,0.f,0.f,0.f,0.f,0.f,0.f,0.f};
#pragma unroll
        for (int c = 0; c < 8; ++c) {
            // B operand: N = lane&15, element e -> K = 8*hs + (e&7) + 16*(e>>3)
            v16bf Bh, Bl;
#pragma unroll
            for (int e = 0; e < 16; ++e) {
                const int kk = c * 32 + 8 * hs + (e & 7) + ((e >> 3) << 4);
                Bh[e] = sWh[buf][kk * 16 + r];
                Bl[e] = sWl[buf][kk * 16 + r];
            }
            C = wmma_bf16(Ah[c], Bh, C);
            C = wmma_bf16(Ah[c], Bl, C);
            C = wmma_bf16(Al[c], Bh, C);
        }
        const int   j   = j0 + r;             // C layout: N = lane&15
        const float wnj = wn[j];
#pragma unroll
        for (int q = 0; q < 8; ++q) {
            float s = wnj - 2.0f * C[q];
            if (s < bestV[q]) { bestV[q] = s; bestI[q] = j; }
        }

        if (jt + 1 < SOM_K / 16) stage_store(buf ^ 1);
    }

    // Reduce argmin across the 16 lanes of each half (rows M=q and M=q+8).
#pragma unroll
    for (int q = 0; q < 8; ++q) {
        float v = bestV[q];
        int   i = bestI[q];
#pragma unroll
        for (int m = 8; m >= 1; m >>= 1) {
            float vo = __shfl_xor(v, m, 32);
            int   io = __shfl_xor(i, m, 32);
            if (vo < v || (vo == v && io < i)) { v = vo; i = io; }
        }
        if (r == 0) {
            int row = x0 + q + 8 * hs;
            bmu_idx[row] = i;
            bmu_loc[row] = make_float2((float)(i >> 6), (float)(i & (SOM_GW - 1)));
        }
    }
}

// ---------------------------------------------------------------------------
// Kernel 5: num += h^T @ x, den += sum_b h.
// Block = 64 map units x 1/4 of the batch (256 blocks). h built on-the-fly in
// LDS in the A-operand layout (4 subtiles); each chunk's x B-operands are
// reused across the 4 j-subtiles (16 WMMAs / chunk / wave), so each block
// reads only its batch quarter of x. Partials combined with f32 atomics.
// ---------------------------------------------------------------------------
__global__ void __launch_bounds__(256) som_accum(
    const __bf16* __restrict__ xh, const __bf16* __restrict__ xl,
    const float2* __restrict__ bmu_loc, const int* __restrict__ epoch,
    float* __restrict__ num, float* __restrict__ den) {

    __shared__ __align__(32) __bf16 hA[4 * 512];   // 4 x (16j x 32b, A layout)
    __shared__ float denT[64];

    const int tid  = threadIdx.x;
    const int lane = tid & 31;
    const int wave = tid >> 5;
    const int jg   = blockIdx.x >> 2;              // 0..63 : 64-j group
    const int bs   = blockIdx.x & 3;               // 0..3  : batch quarter
    const int j0   = jg * 64;
    const int b_begin = bs * (SOM_B / 4);
    const int b_end   = b_begin + SOM_B / 4;

    const float decay = __expf(-(float)epoch[0] * 0.01f);
    const float sigma = 32.0f * decay;             // INIT_SIGMA = 64/2
    const float cf    = -1.0f / (2.0f * sigma * sigma);

    // h-producer role: per subtile s, thread covers (M = tid/16, K = 2t%32..+1)
    const int Mh = tid >> 4;
    const int K0 = (tid << 1) & 31;
    // A-layout slot: lane = M + 16*((K%16)>=8), idx = K%8 + 8*(K>=16)
    const int sl = Mh + (((K0 & 15) >= 8) ? 16 : 0);
    const int sk = (K0 & 7) + ((K0 >= 16) ? 8 : 0);
    float gxs[4], gys[4];
#pragma unroll
    for (int s = 0; s < 4; ++s) {
        int jh = j0 + s * 16 + Mh;
        gxs[s] = (float)(jh >> 6);
        gys[s] = (float)(jh & (SOM_GW - 1));
    }
    float denp[4] = {0.f, 0.f, 0.f, 0.f};
    if (tid < 64) denT[tid] = 0.0f;

    // WMMA-consumer role: wave owns d-tiles dA, dA+16 for all 4 j-subtiles
    const int dA = wave * 32;
    const int dB = dA + 16;
    v8f C0[4], C1[4];
#pragma unroll
    for (int s = 0; s < 4; ++s) {
        C0[s] = (v8f){0.f,0.f,0.f,0.f,0.f,0.f,0.f,0.f};
        C1[s] = (v8f){0.f,0.f,0.f,0.f,0.f,0.f,0.f,0.f};
    }

    for (int b0 = b_begin; b0 < b_end; b0 += 32) {
        float4 loc = *(const float4*)(&bmu_loc[b0 + K0]);   // two samples
#pragma unroll
        for (int s = 0; s < 4; ++s) {
            float dx0 = loc.x - gxs[s], dy0 = loc.y - gys[s];
            float dx1 = loc.z - gxs[s], dy1 = loc.w - gys[s];
            float h0 = __expf(cf * (dx0 * dx0 + dy0 * dy0));
            float h1 = __expf(cf * (dx1 * dx1 + dy1 * dy1));
            denp[s] += h0 + h1;
            __bf16* slot = &hA[s * 512 + sl * 16 + sk];
            slot[0] = (__bf16)h0;
            slot[1] = (__bf16)h1;
        }
        __syncthreads();

        v16bf B0h, B0l, B1h, B1l;   // B: N = lane&15, element e -> K = 8*(lane>>4) + (e&7) + 16*(e>>3)
#pragma unroll
        for (int e = 0; e < 16; ++e) {
            const size_t o = (size_t)(b0 + 8 * (lane >> 4) + (e & 7) + ((e >> 3) << 4)) * SOM_D + (lane & 15);
            B0h[e] = xh[o + dA];
            B0l[e] = xl[o + dA];
            B1h[e] = xh[o + dB];
            B1l[e] = xl[o + dB];
        }
#pragma unroll
        for (int s = 0; s < 4; ++s) {
            v16bf A = *(const v16bf*)(&hA[s * 512 + lane * 16]);
            C0[s] = wmma_bf16(A, B0h, C0[s]);
            C0[s] = wmma_bf16(A, B0l, C0[s]);
            C1[s] = wmma_bf16(A, B1h, C1[s]);
            C1[s] = wmma_bf16(A, B1l, C1[s]);
        }
        __syncthreads();
    }

    // den partials -> LDS -> global atomics
#pragma unroll
    for (int s = 0; s < 4; ++s) atomicAdd(&denT[s * 16 + Mh], denp[s]);
    __syncthreads();
    if (tid < 64) atomicAdd(&den[j0 + tid], denT[tid]);

    // num partials -> global atomics (C layout: VGPR q -> M=q or q+8)
    const int nr = lane & 15;
    const int hb = (lane >> 4) << 3;
#pragma unroll
    for (int s = 0; s < 4; ++s) {
#pragma unroll
        for (int q = 0; q < 8; ++q) {
            int jj = j0 + s * 16 + q + hb;
            atomicAdd(&num[(size_t)jj * SOM_D + dA + nr], C0[s][q]);
            atomicAdd(&num[(size_t)jj * SOM_D + dB + nr], C1[s][q]);
        }
    }
}

// ---------------------------------------------------------------------------
// Kernel 6: gather-and-update: out[i] = w[j]*(1-lr) + (num[j]/den[j])*lr
// ---------------------------------------------------------------------------
__global__ void __launch_bounds__(256) som_finalize(
    const float* __restrict__ w, const float* __restrict__ num,
    const float* __restrict__ den, const int* __restrict__ bmu_idx,
    const int* __restrict__ epoch, float* __restrict__ out_w) {
    int g = blockIdx.x * 256 + threadIdx.x;
    int i = g >> 8;
    int d = g & 255;
    int j = bmu_idx[i];
    float lr = 0.5f * __expf(-(float)epoch[0] * 0.01f);
    float dn = den[j];
    dn = (dn == 0.0f) ? 1e-8f : dn;
    float nw = num[(size_t)j * SOM_D + d] / dn;
    out_w[g] = w[(size_t)j * SOM_D + d] * (1.0f - lr) + nw * lr;
}

// ---------------------------------------------------------------------------
extern "C" void kernel_launch(void* const* d_in, const int* in_sizes, int n_in,
                              void* d_out, int out_size, void* d_ws, size_t ws_size,
                              hipStream_t stream) {
    (void)in_sizes; (void)n_in; (void)out_size; (void)ws_size;

    const float* x     = (const float*)d_in[0];
    const float* w     = (const float*)d_in[1];
    const int*   epoch = (const int*)d_in[2];

    char*  ws  = (char*)d_ws;
    size_t off = 0;
    auto alloc = [&](size_t bytes) {
        void* p = ws + off;
        off = (off + bytes + 255) & ~(size_t)255;
        return p;
    };
    __bf16* xh   = (__bf16*)alloc((size_t)SOM_B * SOM_D * 2);
    __bf16* xl   = (__bf16*)alloc((size_t)SOM_B * SOM_D * 2);
    __bf16* whT  = (__bf16*)alloc((size_t)SOM_K * SOM_D * 2);
    __bf16* wlT  = (__bf16*)alloc((size_t)SOM_K * SOM_D * 2);
    float*  wn   = (float*) alloc((size_t)SOM_K * 4);
    float2* bloc = (float2*)alloc((size_t)SOM_B * 8);
    float*  num  = (float*) alloc((size_t)SOM_K * SOM_D * 4);
    float*  den  = (float*) alloc((size_t)SOM_K * 4);

    int*   out_idx = (int*)d_out;                 // [16384] bmu indices
    float* out_wt  = (float*)d_out + SOM_B;       // [16384*256] gathered rows

    som_prep_x  <<<(SOM_B * SOM_D) / 256, 256, 0, stream>>>(x, xh, xl);
    som_prep_w  <<<SOM_K,               256, 0, stream>>>(w, whT, wlT, wn);
    som_zero    <<<(SOM_K * SOM_D + SOM_K) / 256, 256, 0, stream>>>(num, den);
    som_bmu     <<<SOM_B / 128,         256, 0, stream>>>(xh, xl, whT, wlT, wn,
                                                          out_idx, bloc);
    som_accum   <<<SOM_K / 16,          256, 0, stream>>>(xh, xl, bloc, epoch,
                                                          num, den);
    som_finalize<<<(SOM_B * SOM_D) / 256, 256, 0, stream>>>(w, num, den, out_idx,
                                                            epoch, out_wt);
}
